// Router_352187318549
// MI455X (gfx1250) — compile-verified, matrix operand order checked by reading, not run on hardware
//
#include <hip/hip_runtime.h>
#include <hip/hip_bf16.h>

typedef __attribute__((ext_vector_type(16))) __bf16 v16bf;
typedef __attribute__((ext_vector_type(8)))  __bf16 v8bf;
typedef __attribute__((ext_vector_type(8)))  float  v8f;
typedef __attribute__((ext_vector_type(4)))  float  v4f;

#define T_TOKENS 8192
#define D_DIM    4096
#define E_EXP    64
#define TOPK     8
#define WAVES_PB 4
#define ROWS_PW  16
#define KSPLIT   4
#define KCHUNK   (D_DIM / KSPLIT)          // 1024
#define KITERS   (KCHUNK / 32)             // 32 wmma k-steps per chunk
#define B_CHUNK_ELEMS (KCHUNK * E_EXP)     // 65536 bf16 = 128 KB

// ---------------------------------------------------------------------------
// Pre-pass: repack kernel_DE (fp32 [D,E] row-major) into bf16 WMMA-B layout,
// grouped for conflict-free 16B-stride LDS reads:
//   bws[tile*512 + grp*256 + lane*8 + j]
//   tile = kb*4 + nt ; half = lane>>4
//   K = kb*32 + half*8 + grp*16 + j ; N = nt*16 + (lane&15)
// ---------------------------------------------------------------------------
__global__ void pack_b_kernel(const float* __restrict__ kernel_DE,
                              __bf16* __restrict__ bws) {
    int tid = blockIdx.x * blockDim.x + threadIdx.x;
    if (tid >= D_DIM * E_EXP) return;
    int j    = tid & 7;
    int lane = (tid >> 3) & 31;
    int grp  = (tid >> 8) & 1;
    int tile = tid >> 9;          // kb*4 + nt
    int nt   = tile & 3;
    int kb   = tile >> 2;
    int half = lane >> 4;
    int K = kb * 32 + half * 8 + grp * 16 + j;
    int N = nt * 16 + (lane & 15);
    bws[tid] = (__bf16)kernel_DE[K * E_EXP + N];
}

// ---------------------------------------------------------------------------
// GEMM partial kernel. grid = (row-blocks, KSPLIT). Block = 4 waves.
// B chunk staged in 128 KB LDS; A uses a 2-deep register pipeline so every
// cvt/WMMA consumes loads issued >= 2 iterations earlier.
// ---------------------------------------------------------------------------
struct ARaw { v4f f0, f1, f2, f3; };

__device__ __forceinline__ ARaw load_a_raw(const float* ak) {
    ARaw r;
    r.f0 = *(const v4f*)(ak +  0);
    r.f1 = *(const v4f*)(ak +  4);
    r.f2 = *(const v4f*)(ak + 16);
    r.f3 = *(const v4f*)(ak + 20);
    return r;
}

__device__ __forceinline__ v16bf cvt_a(const ARaw& r) {
    v16bf a;
    #pragma unroll
    for (int i = 0; i < 4; ++i) {
        a[i]      = (__bf16)r.f0[i];
        a[4 + i]  = (__bf16)r.f1[i];
        a[8 + i]  = (__bf16)r.f2[i];
        a[12 + i] = (__bf16)r.f3[i];
    }
    return a;
}

union BV { v16bf v; struct { v8bf lo, hi; } h; };

__device__ __forceinline__ void k_step(const __bf16* ldsB, int kb, int lane,
                                       const ARaw& fa,
                                       v8f& c0, v8f& c1, v8f& c2, v8f& c3) {
    const __bf16* lt = ldsB + kb * 4 * 512 + lane * 8;
    BV b0, b1, b2, b3;
    b0.h.lo = *(const v8bf*)(lt + 0 * 512);
    b0.h.hi = *(const v8bf*)(lt + 0 * 512 + 256);
    b1.h.lo = *(const v8bf*)(lt + 1 * 512);
    b1.h.hi = *(const v8bf*)(lt + 1 * 512 + 256);
    b2.h.lo = *(const v8bf*)(lt + 2 * 512);
    b2.h.hi = *(const v8bf*)(lt + 2 * 512 + 256);
    b3.h.lo = *(const v8bf*)(lt + 3 * 512);
    b3.h.hi = *(const v8bf*)(lt + 3 * 512 + 256);

    v16bf a = cvt_a(fa);

    c0 = __builtin_amdgcn_wmma_f32_16x16x32_bf16(false, a, false, b0.v, (short)0, c0, false, false);
    c1 = __builtin_amdgcn_wmma_f32_16x16x32_bf16(false, a, false, b1.v, (short)0, c1, false, false);
    c2 = __builtin_amdgcn_wmma_f32_16x16x32_bf16(false, a, false, b2.v, (short)0, c2, false, false);
    c3 = __builtin_amdgcn_wmma_f32_16x16x32_bf16(false, a, false, b3.v, (short)0, c3, false, false);
}

__global__ __launch_bounds__(WAVES_PB * 32)
void gemm_partial_kernel(const float* __restrict__ x,
                         const __bf16* __restrict__ bws,
                         float* __restrict__ part) {
    __shared__ __align__(16) __bf16 ldsB[B_CHUNK_ELEMS];   // 128 KB

    const int lane = threadIdx.x & 31;
    const int wave = threadIdx.x >> 5;
    const int l15  = lane & 15;
    const int half = lane >> 4;
    const int rowBase = (blockIdx.x * WAVES_PB + wave) * ROWS_PW;
    const int kc      = blockIdx.y;
    const int kbase   = kc * KCHUNK;

    // ---- cooperative stage of this block's B chunk into LDS ----
    {
        const uint4* src = (const uint4*)(bws + (size_t)(kbase / 32) * 4 * 512);
        uint4* dst = (uint4*)ldsB;
        #pragma unroll 4
        for (int i = threadIdx.x; i < B_CHUNK_ELEMS / 8; i += WAVES_PB * 32)
            dst[i] = src[i];
    }
    __syncthreads();

    v8f c0 = {}, c1 = {}, c2 = {}, c3 = {};

    // A layout (16-bit 16x32): lanes 0-15 rows M, K {0..7,16..23};
    // lanes 16-31 same rows, K {8..15,24..31}.
    const float* arow = x + (size_t)(rowBase + l15) * D_DIM + kbase + half * 8;

    // ---- 2-deep A pipeline ----
    ARaw fa0 = load_a_raw(arow);
    ARaw fa1 = load_a_raw(arow + 32);

    for (int kb = 0; kb < KITERS - 2; ++kb) {
        ARaw fa2 = load_a_raw(arow + (kb + 2) * 32);   // 2 iterations ahead
        k_step(ldsB, kb, lane, fa0, c0, c1, c2, c3);
        fa0 = fa1;
        fa1 = fa2;
    }
    k_step(ldsB, KITERS - 2, lane, fa0, c0, c1, c2, c3);
    k_step(ldsB, KITERS - 1, lane, fa1, c0, c1, c2, c3);

    // C layout: lane -> col N%16, rows M = half*8 + v. Store partial tile.
    float* pb = part + ((size_t)kc * T_TOKENS + rowBase) * E_EXP;
    #pragma unroll
    for (int v = 0; v < 8; ++v) {
        int r = half * 8 + v;
        pb[(size_t)r * E_EXP +  0 + l15] = c0[v];
        pb[(size_t)r * E_EXP + 16 + l15] = c1[v];
        pb[(size_t)r * E_EXP + 32 + l15] = c2[v];
        pb[(size_t)r * E_EXP + 48 + l15] = c3[v];
    }
}

// ---------------------------------------------------------------------------
// Reduction + top-8 + softmax. Block = 256 threads handles 64 rows.
// ---------------------------------------------------------------------------
__global__ __launch_bounds__(256)
void reduce_topk_kernel(const float* __restrict__ part,
                        float* __restrict__ out) {
    __shared__ float lds[64][E_EXP + 1];   // +1 pad: conflict-free row scans

    const int t = threadIdx.x;
    const int rowBlk = blockIdx.x * 64;
    const size_t stride = (size_t)T_TOKENS * E_EXP;

    #pragma unroll
    for (int i = 0; i < 16; ++i) {
        int idx = i * 256 + t;               // 0 .. 4095
        int r = idx >> 6;
        int e = idx & 63;
        size_t off = (size_t)(rowBlk + r) * E_EXP + e;
        float s = part[off] + part[stride + off]
                + part[2 * stride + off] + part[3 * stride + off];
        lds[r][e] = s;
    }
    __syncthreads();

    if (t < 64) {
        const int rowG = rowBlk + t;
        float vals[TOPK];
        int   idx[TOPK];
        #pragma unroll
        for (int j = 0; j < TOPK; ++j) {
            float best = -3.4e38f;
            int   bi   = 0;
            for (int e = 0; e < E_EXP; ++e) {
                float v = lds[t][e];
                if (v > best) { best = v; bi = e; }
            }
            lds[t][bi] = -3.4e38f;           // mask out; ties -> lowest index
            vals[j] = best;
            idx[j]  = bi;
        }
        float m = vals[0];                   // descending; vals[0] == max
        float w[TOPK];
        float s = 0.0f;
        #pragma unroll
        for (int j = 0; j < TOPK; ++j) { w[j] = __expf(vals[j] - m); s += w[j]; }
        float inv = 1.0f / s;
        #pragma unroll
        for (int j = 0; j < TOPK; ++j) {
            out[(size_t)rowG * TOPK + j] = w[j] * inv;
            out[(size_t)T_TOKENS * TOPK + (size_t)rowG * TOPK + j] = (float)idx[j];
        }
    }
}

extern "C" void kernel_launch(void* const* d_in, const int* in_sizes, int n_in,
                              void* d_out, int out_size, void* d_ws, size_t ws_size,
                              hipStream_t stream) {
    const float* x        = (const float*)d_in[0];   // [8192, 4096] fp32
    const float* kernelDE = (const float*)d_in[1];   // [4096, 64]  fp32
    float* out = (float*)d_out;                      // weights[T*8] ++ indices[T*8]

    // workspace layout: [0, 512KB) packed bf16 B ; [512KB, 512KB+8MB) partials
    __bf16* bws  = (__bf16*)d_ws;
    float*  part = (float*)((char*)d_ws + (size_t)D_DIM * E_EXP * sizeof(__bf16));

    pack_b_kernel<<<(D_DIM * E_EXP + 255) / 256, 256, 0, stream>>>(kernelDE, bws);

    dim3 grid(T_TOKENS / (WAVES_PB * ROWS_PW), KSPLIT);   // 128 x 4
    gemm_partial_kernel<<<grid, WAVES_PB * 32, 0, stream>>>(x, bws, part);

    reduce_topk_kernel<<<T_TOKENS / 64, 256, 0, stream>>>(part, out);
}
